// AlignerOT_91259465106078
// MI455X (gfx1250) — matrix operand order repaired and checked by reading, not compiled, not run on hardware
//
#include <hip/hip_runtime.h>
#include <hip/hip_bf16.h>

#define DD 512
#define NPROB 256
#define ITERS 50

typedef __attribute__((ext_vector_type(16))) __bf16 v16bf;
typedef __attribute__((ext_vector_type(8)))  float  v8f;

// ---------------------------------------------------------------------------
// float -> bf16 (round-to-nearest-even), also returning the float value of
// the truncated hi part so we can form the residual for the bf16x3 split.
// ---------------------------------------------------------------------------
__device__ __forceinline__ __bf16 f32_to_bf16_rne(float f, float* hi_as_f32) {
    unsigned u = __builtin_bit_cast(unsigned, f);
    unsigned r = u + 0x7FFFu + ((u >> 16) & 1u);
    unsigned short h = (unsigned short)(r >> 16);
    *hi_as_f32 = __builtin_bit_cast(float, (unsigned)h << 16);
    return __builtin_bit_cast(__bf16, h);
}

// ---------------------------------------------------------------------------
// Kernel 1: per-problem log-domain Sinkhorn. One block per n (256 blocks),
// 256 threads, each thread owns rows {tid, tid+256}. Everything in LDS.
// Stores f10 = f/reg and g10 = g/reg  (reg = 0.1).
// cost/reg = 3000*(s_i - t_j)^2 recomputed on the fly (never materialized).
// ---------------------------------------------------------------------------
__global__ __launch_bounds__(256) void sinkhorn_fg_kernel(
    const float* __restrict__ X, const float* __restrict__ Y,
    float* __restrict__ F10, float* __restrict__ G10)
{
    __shared__ float s_s[DD];   // source row values
    __shared__ float s_t[DD];   // target row values
    __shared__ float s_f[DD];   // f10
    __shared__ float s_g[DD];   // g10

    const int n   = blockIdx.x;
    const int tid = threadIdx.x;

    for (int k = tid; k < DD; k += 256) {
        s_s[k] = X[n * DD + k];
        s_t[k] = Y[n * DD + k];
        s_f[k] = 0.0f;
        s_g[k] = 0.0f;
    }
    __syncthreads();

    const float la = -6.238324625039508f;   // log(1/512) == lb
    const int i0 = tid, i1 = tid + 256;

    for (int it = 0; it < ITERS; ++it) {
        // ---- f update: f10_i = la - logsumexp_j( g10_j - 3000*(s_i-t_j)^2 )
        float f0v, f1v;
        {
            const float a = s_s[i0], b = s_s[i1];
            float m0 = -3.0e38f, m1 = -3.0e38f;
            #pragma unroll 4
            for (int j = 0; j < DD; ++j) {
                const float tj = s_t[j], gj = s_g[j];
                float d0 = a - tj; float x0 = fmaf(d0 * -3000.0f, d0, gj);
                float d1 = b - tj; float x1 = fmaf(d1 * -3000.0f, d1, gj);
                m0 = fmaxf(m0, x0); m1 = fmaxf(m1, x1);
            }
            float sum0 = 0.0f, sum1 = 0.0f;
            #pragma unroll 4
            for (int j = 0; j < DD; ++j) {
                const float tj = s_t[j], gj = s_g[j];
                float d0 = a - tj; float x0 = fmaf(d0 * -3000.0f, d0, gj);
                float d1 = b - tj; float x1 = fmaf(d1 * -3000.0f, d1, gj);
                sum0 += __expf(x0 - m0); sum1 += __expf(x1 - m1);
            }
            f0v = la - (m0 + __logf(sum0));
            f1v = la - (m1 + __logf(sum1));
        }
        s_f[i0] = f0v; s_f[i1] = f1v;      // nobody reads s_f during f pass
        __syncthreads();                    // all f10 visible before g pass

        // ---- g update: g10_j = la - logsumexp_i( f10_i - 3000*(s_i-t_j)^2 )
        float g0v, g1v;
        {
            const float a = s_t[i0], b = s_t[i1];
            float m0 = -3.0e38f, m1 = -3.0e38f;
            #pragma unroll 4
            for (int i = 0; i < DD; ++i) {
                const float si = s_s[i], fi = s_f[i];
                float d0 = a - si; float x0 = fmaf(d0 * -3000.0f, d0, fi);
                float d1 = b - si; float x1 = fmaf(d1 * -3000.0f, d1, fi);
                m0 = fmaxf(m0, x0); m1 = fmaxf(m1, x1);
            }
            float sum0 = 0.0f, sum1 = 0.0f;
            #pragma unroll 4
            for (int i = 0; i < DD; ++i) {
                const float si = s_s[i], fi = s_f[i];
                float d0 = a - si; float x0 = fmaf(d0 * -3000.0f, d0, fi);
                float d1 = b - si; float x1 = fmaf(d1 * -3000.0f, d1, fi);
                sum0 += __expf(x0 - m0); sum1 += __expf(x1 - m1);
            }
            g0v = la - (m0 + __logf(sum0));
            g1v = la - (m1 + __logf(sum1));
        }
        s_g[i0] = g0v; s_g[i1] = g1v;
        __syncthreads();                    // all g10 visible before next iter
    }

    F10[n * DD + i0] = s_f[i0];
    F10[n * DD + i1] = s_f[i1];
    G10[n * DD + i0] = s_g[i0];
    G10[n * DD + i1] = s_g[i1];
}

// ---------------------------------------------------------------------------
// Kernel 2: ot_matrix[i][j] = 600 * sum_n exp(f10+g10-3000*(s-t)^2) + delta.
// 64 blocks, each owns a 64x64 tile; loops n, staging the 4x64 floats it
// needs per n in LDS; 4x4 accumulators per thread. 600 = D*SCALE/N.
// ---------------------------------------------------------------------------
__global__ __launch_bounds__(256) void accum_ot_kernel(
    const float* __restrict__ X, const float* __restrict__ Y,
    const float* __restrict__ F10, const float* __restrict__ G10,
    const float* __restrict__ delta, float* __restrict__ OT)
{
    __shared__ float sh[4][64];            // s, f10, t, g10 for current n

    const int ti = (blockIdx.x >> 3) * 64; // i-tile origin
    const int tj = (blockIdx.x & 7) * 64;  // j-tile origin
    const int tx = threadIdx.x & 15;
    const int ty = threadIdx.x >> 4;

    float acc[4][4] = {};

    for (int n = 0; n < NPROB; ++n) {
        __syncthreads();                   // previous-n reads done
        {
            const int sel = threadIdx.x >> 6;   // 0..3
            const int l   = threadIdx.x & 63;
            const float* src = (sel == 0) ? (X   + n * DD + ti + l)
                             : (sel == 1) ? (F10 + n * DD + ti + l)
                             : (sel == 2) ? (Y   + n * DD + tj + l)
                                          : (G10 + n * DD + tj + l);
            sh[sel][l] = *src;
        }
        __syncthreads();

        #pragma unroll
        for (int ii = 0; ii < 4; ++ii) {
            const float si = sh[0][ty + 16 * ii];
            const float fi = sh[1][ty + 16 * ii];
            #pragma unroll
            for (int jj = 0; jj < 4; ++jj) {
                const float d   = si - sh[2][tx + 16 * jj];
                const float arg = fmaf(d * -3000.0f, d, fi + sh[3][tx + 16 * jj]);
                acc[ii][jj] += __expf(arg);
            }
        }
    }

    #pragma unroll
    for (int ii = 0; ii < 4; ++ii) {
        const int i = ti + ty + 16 * ii;
        #pragma unroll
        for (int jj = 0; jj < 4; ++jj) {
            const int j = tj + tx + 16 * jj;
            OT[i * DD + j] = acc[ii][jj] * 600.0f + delta[i * DD + j];
        }
    }
}

// ---------------------------------------------------------------------------
// Kernel 3: out = src @ ot_matrix via v_wmma_f32_16x16x32_bf16, bf16x3 split
// (hi*hi + hi*lo + lo*hi) for ~f32 accuracy. One 16x16 tile per wave32.
// M=256, N=512, K=512 -> 512 tiles, 8 waves/block, 64 blocks.
// ---------------------------------------------------------------------------
__global__ __launch_bounds__(256) void gemm_wmma_kernel(
    const float* __restrict__ X, const float* __restrict__ OT,
    float* __restrict__ out)
{
    const int wave = threadIdx.x >> 5;
    const int lane = threadIdx.x & 31;
    const int tile = blockIdx.x * 8 + wave;
    const int tm = tile >> 5;              // 0..15  (M tiles)
    const int tn = tile & 31;              // 0..31  (N tiles)

    const int row = tm * 16 + (lane & 15); // A: lane = M row
    const int col = tn * 16 + (lane & 15); // B/C/D: lane = N column
    const int kbA = (lane >> 4) * 8;       // A half-wave K base
    const int kbB = (lane >> 4) * 8;       // B half-wave K base

    v8f c = {};
    for (int k0 = 0; k0 < DD; k0 += 32) {
        v16bf ahi, alo, bhi, blo;
        #pragma unroll
        for (int e = 0; e < 16; ++e) {
            // A element e -> K = kbA + e (+8 for upper VGPR quad)
            const int ka = k0 + kbA + e + ((e >= 8) ? 8 : 0);
            const float av = X[row * DD + ka];
            float ah;
            ahi[e] = f32_to_bf16_rne(av, &ah);
            float dummy;
            alo[e] = f32_to_bf16_rne(av - ah, &dummy);
            // B element e -> K = kbB + e (+8 for upper VGPR quad)
            const int kb = k0 + kbB + e + ((e >= 8) ? 8 : 0);
            const float bv = OT[kb * DD + col];
            float bh;
            bhi[e] = f32_to_bf16_rne(bv, &bh);
            blo[e] = f32_to_bf16_rne(bv - bh, &dummy);
        }
        c = __builtin_amdgcn_wmma_f32_16x16x32_bf16(false, ahi, false, bhi,
                                                    (short)0, c, false, false);
        c = __builtin_amdgcn_wmma_f32_16x16x32_bf16(false, ahi, false, blo,
                                                    (short)0, c, false, false);
        c = __builtin_amdgcn_wmma_f32_16x16x32_bf16(false, alo, false, bhi,
                                                    (short)0, c, false, false);
    }

    // C/D layout: VGPR v -> M = v (+8 for upper half-wave), lane&15 -> N
    const int mb = tm * 16 + ((lane >> 4) ? 8 : 0);
    #pragma unroll
    for (int v = 0; v < 8; ++v)
        out[(mb + v) * DD + col] = c[v];
}

// ---------------------------------------------------------------------------
extern "C" void kernel_launch(void* const* d_in, const int* in_sizes, int n_in,
                              void* d_out, int out_size, void* d_ws, size_t ws_size,
                              hipStream_t stream) {
    (void)in_sizes; (void)n_in; (void)out_size; (void)ws_size;
    const float* X     = (const float*)d_in[0];   // [2,128,512] -> [256,512]
    const float* Y     = (const float*)d_in[1];   // [2,128,512] -> [256,512]
    const float* delta = (const float*)d_in[2];   // [512,512]
    float* out = (float*)d_out;                   // [256,512]

    char* ws = (char*)d_ws;
    float* F10 = (float*)(ws);                        // 256*512 f32 = 512 KB
    float* G10 = (float*)(ws + (size_t)NPROB * DD * 4);        // 512 KB
    float* OT  = (float*)(ws + (size_t)2 * NPROB * DD * 4);    // 512*512 f32 = 1 MB

    sinkhorn_fg_kernel<<<NPROB, 256, 0, stream>>>(X, Y, F10, G10);
    accum_ot_kernel<<<64, 256, 0, stream>>>(X, Y, F10, G10, delta, OT);
    gemm_wmma_kernel<<<64, 256, 0, stream>>>(X, OT, out);
}
